// MSA_5420248727621
// MI455X (gfx1250) — matrix operand order repaired and checked by reading, not compiled, not run on hardware
//
#include <hip/hip_runtime.h>

// Problem constants (match reference)
#define BB 8
#define NN 2048
#define DD 768

typedef __attribute__((ext_vector_type(16))) _Float16 v16h;
typedef __attribute__((ext_vector_type(8)))  _Float16 v8h;
typedef __attribute__((ext_vector_type(8)))  float    v8f;

// ---------------------------------------------------------------------------
// CDNA5 async global->LDS copy (ASYNCcnt-tracked). Per-lane 16B transfer:
//   LDS[vdst_lane] = MEM[vaddr_lane]   (GLOBAL_LOAD_ASYNC_TO_LDS_B128)
// LDS address = low 32 bits of the flat shared pointer (aperture bits are all
// in addr[63:32] per the ISA aperture decomposition).
// ---------------------------------------------------------------------------
__device__ __forceinline__ void async_b128(uint32_t lds_addr, const void* gsrc) {
  asm volatile("global_load_async_to_lds_b128 %0, %1, off"
               :: "v"(lds_addr), "v"(gsrc) : "memory");
}
__device__ __forceinline__ void wait_async0() {
  asm volatile("s_wait_asynccnt 0x0" ::: "memory");
}

// Stage one contiguous `bytes` block (multiple of 4096 here) from global to
// LDS with all 256 threads, then drain this wave's ASYNCcnt.
__device__ __forceinline__ void stage_tile_async(void* lds_base,
                                                 const void* g_base,
                                                 int bytes) {
  const uint32_t l0 = (uint32_t)(uintptr_t)lds_base;
  const char* g = (const char*)g_base;
  for (int off = (int)threadIdx.x * 16; off < bytes; off += 256 * 16)
    async_b128(l0 + (uint32_t)off, g + off);
  wait_async0();
}

// ---------------------------------------------------------------------------
// WMMA helper: D = A(16x32 f16) * B(32x16 f16) + C(16x16 f32)
// ---------------------------------------------------------------------------
__device__ __forceinline__ v8f wmma_f16f32(v16h a, v16h b, v8f c) {
  return __builtin_amdgcn_wmma_f32_16x16x32_f16(
      /*neg_a=*/false, a, /*neg_b=*/false, b,
      /*c_mod=*/(short)0, c, /*reuse_a=*/false, /*reuse_b=*/false);
}

// A fragment (16x32, f16). Lane L (<16) row M=L holds
// K = k0+{0..15}; lane L+16 holds K = k0+16+{0..15}.
// Works for both global and LDS source pointers (generic addressing).
__device__ __forceinline__ v16h load_a_f16(const _Float16* base,
                                           int ld, int row0, int k0) {
  const int lane = threadIdx.x & 31;
  const _Float16* p = base + (size_t)(row0 + (lane & 15)) * ld
                           + k0 + ((lane >> 4) << 4);
  v8h lo = *(const v8h*)p;          // K = k0 + h*16 + {0..7}
  v8h hi = *(const v8h*)(p + 8);    // K = k0 + h*16 + 8 + {0..7}
  v16h a;
#pragma unroll
  for (int i = 0; i < 8; ++i) { a[i] = lo[i]; a[i + 8] = hi[i]; }
  return a;
}

// A fragment built from an f32 LDS row panel (for attn weights): same K
// pattern, converted f32 -> f16 in registers.
__device__ __forceinline__ v16h load_a_from_lds_f32(const float* S,
                                                    int ld, int k0) {
  const int lane = threadIdx.x & 31;
  const float* p = S + (size_t)(lane & 15) * ld + k0 + ((lane >> 4) << 4);
  v16h a;
#pragma unroll
  for (int i = 0; i < 8; ++i) {
    a[i]     = (_Float16)p[i];
    a[i + 8] = (_Float16)p[i + 8];
  }
  return a;
}

// B fragment (32x16, f16), source stored "column-rows contiguous over K":
// lane owns column n = n0 + (lane&15); K = k0 + (lane>>4)*16 + {0..15}.
__device__ __forceinline__ v16h load_b_f16(const _Float16* __restrict__ base,
                                           int ld, int n0, int k0) {
  const int lane = threadIdx.x & 31;
  const _Float16* p = base + (size_t)(n0 + (lane & 15)) * ld
                           + k0 + ((lane >> 4) << 4);
  return *(const v16h*)p;
}

// ---------------------------------------------------------------------------
// wave32 reductions
// ---------------------------------------------------------------------------
__device__ __forceinline__ float wave_max_f32(float v) {
#pragma unroll
  for (int o = 16; o > 0; o >>= 1) v = fmaxf(v, __shfl_xor(v, o, 32));
  return v;
}
__device__ __forceinline__ float wave_sum_f32(float v) {
#pragma unroll
  for (int o = 16; o > 0; o >>= 1) v += __shfl_xor(v, o, 32);
  return v;
}

// ---------------------------------------------------------------------------
// 1) Fused LayerNorm -> f16.  One block per token row (768 = 3*256).
// ---------------------------------------------------------------------------
__global__ void msa_ln_f16_kernel(const float* __restrict__ x,
                                  const float* __restrict__ gamma,
                                  const float* __restrict__ beta,
                                  _Float16* __restrict__ xn16) {
  const int row = blockIdx.x;               // 0 .. B*N-1
  const int tid = threadIdx.x;              // 0 .. 255
  const float* xr = x + (size_t)row * DD;

  float v0 = xr[tid], v1 = xr[tid + 256], v2 = xr[tid + 512];
  float s  = v0 + v1 + v2;
  float s2 = v0 * v0 + v1 * v1 + v2 * v2;

  __shared__ float sh[16];
  s  = wave_sum_f32(s);
  s2 = wave_sum_f32(s2);
  const int w = tid >> 5, lane = tid & 31;
  if (lane == 0) { sh[w] = s; sh[w + 8] = s2; }
  __syncthreads();
  if (tid == 0) {
    float a = 0.f, b = 0.f;
#pragma unroll
    for (int i = 0; i < 8; ++i) { a += sh[i]; b += sh[i + 8]; }
    sh[0] = a; sh[8] = b;
  }
  __syncthreads();
  const float mean = sh[0] * (1.0f / DD);
  const float var  = sh[8] * (1.0f / DD) - mean * mean;
  const float rstd = rsqrtf(var + 1e-5f);

  _Float16* o = xn16 + (size_t)row * DD;
  o[tid]       = (_Float16)((v0 - mean) * rstd * gamma[tid]       + beta[tid]);
  o[tid + 256] = (_Float16)((v1 - mean) * rstd * gamma[tid + 256] + beta[tid + 256]);
  o[tid + 512] = (_Float16)((v2 - mean) * rstd * gamma[tid + 512] + beta[tid + 512]);
}

// ---------------------------------------------------------------------------
// 2) f32 -> f16 weight convert
// ---------------------------------------------------------------------------
__global__ void msa_cvt_f16_kernel(const float* __restrict__ src,
                                   _Float16* __restrict__ dst, int n) {
  int i = blockIdx.x * 256 + threadIdx.x;
  if (i < n) dst[i] = (_Float16)src[i];
}

// ---------------------------------------------------------------------------
// 3) QKV projection via WMMA.  grid = (B*N/16, D/128, 3), block = 256 (8 waves).
//    The 16x768 activation tile (24 KB, contiguous) is staged once into LDS
//    via async global->LDS (all 8 waves share it); each wave then runs a
//    16x16-tile K-loop of 24 WMMAs with B-fragments streamed from global.
//    z==2 (V) is stored TRANSPOSED: Vt[b][d][n], so the attention P*V GEMM
//    gets contiguous B-fragments.
// ---------------------------------------------------------------------------
__global__ void msa_qkv_wmma_kernel(const _Float16* __restrict__ xn16,
                                    const _Float16* __restrict__ w16,
                                    const float* __restrict__ bq,
                                    const float* __restrict__ bk,
                                    const float* __restrict__ bv,
                                    _Float16* __restrict__ q16,
                                    _Float16* __restrict__ k16,
                                    _Float16* __restrict__ vt16) {
  __shared__ _Float16 xs[16 * DD];                   // 24 KB activation tile

  const int lane = threadIdx.x & 31;
  const int wid  = threadIdx.x >> 5;
  const int m0   = blockIdx.x * 16;                  // flattened token tile
  const int n0   = (blockIdx.y * 8 + wid) * 16;      // output-feature tile
  const int z    = blockIdx.z;                       // 0:q 1:k 2:v

  const _Float16* w   = w16 + (size_t)z * DD * DD;   // [OUT][IN] row-major
  const float* bias   = (z == 0) ? bq : ((z == 1) ? bk : bv);

  // Async-stage the shared A tile (16 consecutive rows -> contiguous 24 KB).
  stage_tile_async(xs, xn16 + (size_t)m0 * DD, 16 * DD * 2);
  __syncthreads();

  v8f acc = {};
  for (int k = 0; k < DD; k += 32) {
    v16h a = load_a_f16(xs, DD, 0, k);               // activations from LDS
    v16h b = load_b_f16(w,  DD, n0, k);              // W row n is B column n
    acc = wmma_f16f32(a, b, acc);
  }

  const int nloc  = n0 + (lane & 15);
  const float bvv = bias[nloc];
  const int mbase = m0 + ((lane >> 4) << 3);

  if (z == 2) {
#pragma unroll
    for (int r = 0; r < 8; ++r) {
      const int m = mbase + r;                       // flattened token index
      const int bi = m >> 11, t = m & (NN - 1);
      vt16[((size_t)bi * DD + nloc) * NN + t] = (_Float16)(acc[r] + bvv);
    }
  } else {
    _Float16* dst = (z == 0) ? q16 : k16;
#pragma unroll
    for (int r = 0; r < 8; ++r) {
      const int m = mbase + r;
      dst[(size_t)m * DD + nloc] = (_Float16)(acc[r] + bvv);
    }
  }
}

// ---------------------------------------------------------------------------
// 4) Attention: one block per (b, 16-query tile); 256 threads / 8 waves.
//    LDS: 24 KB Q tile (async-staged) + 128 KB f32 score panel + rowsums,
//    ~152 KB of the 320 KB/WGP -> scores never touch HBM.
//    Phase 1: S = Q*K^T   (waves split the 128 key tiles; Q frags from LDS)
//    Phase 2: softmax rows in place (wave32 shuffle reductions, deferred 1/sum)
//    Phase 3: O = P*V     (waves split D into 8 x 96-column slabs)
// ---------------------------------------------------------------------------
__global__ void msa_attn_wmma_kernel(const _Float16* __restrict__ q16,
                                     const _Float16* __restrict__ k16,
                                     const _Float16* __restrict__ vt16,
                                     float* __restrict__ out) {
  extern __shared__ char smem_raw[];
  _Float16* Qs = (_Float16*)smem_raw;                 // [16][DD] f16, 24 KB
  float* S     = (float*)(smem_raw + 16 * DD * 2);    // [16][NN] f32, 128 KB
  float* rsum  = S + 16 * NN;                         // [16]

  const int lane = threadIdx.x & 31;
  const int wid  = threadIdx.x >> 5;
  const int b    = blockIdx.x >> 7;            // NN/16 = 128 q-tiles per batch
  const int q0   = (blockIdx.x & 127) * 16;

  const _Float16* Q  = q16  + ((size_t)b * NN + q0) * DD;
  const _Float16* K  = k16  + (size_t)b * NN * DD;
  const _Float16* Vt = vt16 + (size_t)b * DD * NN;

  // Async-stage the Q tile (16 consecutive rows -> contiguous 24 KB).
  stage_tile_async(Qs, Q, 16 * DD * 2);
  __syncthreads();

  // ---- Phase 1: S[16][2048] = Q * K^T ----
  for (int j = 0; j < 16; ++j) {
    const int key0 = (wid * 16 + j) * 16;
    v8f acc = {};
    for (int k = 0; k < DD; k += 32) {
      v16h a = load_a_f16(Qs, DD, 0, k);       // queries from LDS
      v16h bfr = load_b_f16(K, DD, key0, k);   // K row = B column (K^T)
      acc = wmma_f16f32(a, bfr, acc);
    }
    const int mrel = (lane >> 4) << 3;
    const int n    = key0 + (lane & 15);
#pragma unroll
    for (int r = 0; r < 8; ++r) S[(size_t)(mrel + r) * NN + n] = acc[r];
  }
  __syncthreads();

  // ---- Phase 2: softmax over keys, in place; wave wid does rows wid, wid+8 ----
#pragma unroll
  for (int rr = 0; rr < 2; ++rr) {
    const int row = wid + rr * 8;
    float* sr = S + (size_t)row * NN;
    float mx = -3.0e38f;
    for (int kk = lane; kk < NN; kk += 32) mx = fmaxf(mx, sr[kk]);
    mx = wave_max_f32(mx);
    float sum = 0.f;
    for (int kk = lane; kk < NN; kk += 32) {
      float e = __expf(sr[kk] - mx);
      sr[kk] = e;
      sum += e;
    }
    sum = wave_sum_f32(sum);
    if (lane == 0) rsum[row] = sum;
  }
  __syncthreads();

  // ---- Phase 3: O = P * V.  Wave wid owns d-columns [wid*96, wid*96+96). ----
  v8f acc[6] = {};
  for (int k0 = 0; k0 < NN; k0 += 32) {
    v16h a = load_a_from_lds_f32(S, NN, k0);   // attn weights f32->f16
#pragma unroll
    for (int j = 0; j < 6; ++j) {
      const int d0 = wid * 96 + j * 16;
      v16h bfr = load_b_f16(Vt, NN, d0, k0);   // Vt row d contiguous over keys
      acc[j] = wmma_f16f32(a, bfr, acc[j]);
    }
  }

  const int mrel = (lane >> 4) << 3;
#pragma unroll
  for (int r = 0; r < 8; ++r) {
    const float inv = 1.0f / rsum[mrel + r];
    const size_t orow = ((size_t)b * NN + q0 + mrel + r) * DD;
#pragma unroll
    for (int j = 0; j < 6; ++j) {
      const int d = wid * 96 + j * 16 + (lane & 15);
      out[orow + d] = acc[j][r] * inv;
    }
  }
}

// ---------------------------------------------------------------------------
// Host launcher
// ---------------------------------------------------------------------------
extern "C" void kernel_launch(void* const* d_in, const int* in_sizes, int n_in,
                              void* d_out, int out_size, void* d_ws, size_t ws_size,
                              hipStream_t stream) {
  const float* x     = (const float*)d_in[0];
  const float* gamma = (const float*)d_in[1];
  const float* beta  = (const float*)d_in[2];
  const float* wq    = (const float*)d_in[3];
  const float* bq    = (const float*)d_in[4];
  const float* wk    = (const float*)d_in[5];
  const float* bk    = (const float*)d_in[6];
  const float* wv    = (const float*)d_in[7];
  const float* bv    = (const float*)d_in[8];
  float* out = (float*)d_out;

  // Workspace layout (f16 intermediates), ~100 MB total.
  char* ws = (char*)d_ws;
  const size_t SZ_XN = (size_t)BB * NN * DD * sizeof(_Float16);  // 24 MB
  const size_t SZ_W  = (size_t)3 * DD * DD * sizeof(_Float16);   // 3.4 MB
  _Float16* xn16 = (_Float16*)(ws);
  _Float16* w16  = (_Float16*)(ws + SZ_XN);
  _Float16* q16  = (_Float16*)(ws + SZ_XN + SZ_W);
  _Float16* k16  = (_Float16*)(ws + SZ_XN + SZ_W + SZ_XN);
  _Float16* vt16 = (_Float16*)(ws + SZ_XN + SZ_W + 2 * SZ_XN);

  // 1) LayerNorm -> f16
  msa_ln_f16_kernel<<<BB * NN, 256, 0, stream>>>(x, gamma, beta, xn16);

  // 2) Weights -> f16
  const int wn = DD * DD;
  const int wgrid = (wn + 255) / 256;
  msa_cvt_f16_kernel<<<wgrid, 256, 0, stream>>>(wq, w16,          wn);
  msa_cvt_f16_kernel<<<wgrid, 256, 0, stream>>>(wk, w16 +     wn, wn);
  msa_cvt_f16_kernel<<<wgrid, 256, 0, stream>>>(wv, w16 + 2 * wn, wn);

  // 3) QKV projections (V stored transposed)
  msa_qkv_wmma_kernel<<<dim3(BB * NN / 16, DD / 128, 3), 256, 0, stream>>>(
      xn16, w16, bq, bk, bv, q16, k16, vt16);

  // 4) Attention: 24 KB Q tile + 128 KB score panel + rowsums in LDS
  const int SMEM = 16 * DD * 2 + (16 * NN + 16) * (int)sizeof(float);
  hipFuncSetAttribute(reinterpret_cast<const void*>(msa_attn_wmma_kernel),
                      hipFuncAttributeMaxDynamicSharedMemorySize, SMEM);
  msa_attn_wmma_kernel<<<dim3(BB * NN / 16), 256, SMEM, stream>>>(
      q16, k16, vt16, out);
}
